// CHGANSimplified_62577673503583
// MI455X (gfx1250) — matrix operand order repaired and checked by reading, not compiled, not run on hardware
//
#include <hip/hip_runtime.h>

typedef __attribute__((ext_vector_type(16))) __bf16 v16bf;
typedef __attribute__((ext_vector_type(8)))  __bf16 v8bf;
typedef __attribute__((ext_vector_type(8)))  float  v8f;

#define DEVC constexpr int
DEVC Nn = 1024, Dd = 128, Hh = 4, Tt = 12, Bb = 2, DHd = 32;
DEVC ROWS = Bb * Nn * Tt;          // 24576 flattened (b,n,t) rows
DEVC WMAT = Dd * Dd;               // 16384 per weight matrix
DEVC NCH  = Nn / 32;               // 32 key chunks of 32

// ---- CDNA5 async copy: global -> LDS, 16 bytes per lane, tracked by ASYNCcnt
__device__ __forceinline__ void async_g2l_b128(unsigned lds_off, const void* gaddr) {
  asm volatile("global_load_async_to_lds_b128 %0, %1, off"
               :: "v"(lds_off), "v"(gaddr) : "memory");
}
__device__ __forceinline__ void wait_async_le4() {
  asm volatile("s_wait_asynccnt 4" ::: "memory");
}
__device__ __forceinline__ void wait_async_le0() {
  asm volatile("s_wait_asynccnt 0" ::: "memory");
}

// ---------------------------------------------------------------- prep kernels
__global__ __launch_bounds__(256)
void prep_weights(const float* __restrict__ wq, const float* __restrict__ wk,
                  const float* __restrict__ wv, const float* __restrict__ wo,
                  __bf16* __restrict__ q, __bf16* __restrict__ k,
                  __bf16* __restrict__ v, __bf16* __restrict__ o) {
  int i = blockIdx.x * 256 + threadIdx.x;
  if (i >= WMAT) return;
  q[i] = (__bf16)wq[i]; k[i] = (__bf16)wk[i];
  v[i] = (__bf16)wv[i]; o[i] = (__bf16)wo[i];
}

__global__ __launch_bounds__(256)
void prep_enh(const float* __restrict__ x, const float* __restrict__ emb,
              __bf16* __restrict__ enh) {
  int i = blockIdx.x * 256 + threadIdx.x;           // < ROWS*Dd
  int c = i & (Dd - 1);
  int rid = i >> 7;                                 // (b*N + n)*T + t
  int n = (rid / Tt) % Nn;
  int ty = 1 - (n & 1);                             // even n -> type 1
  enh[i] = (__bf16)(x[i] + emb[ty * Dd + c]);
}

__global__ __launch_bounds__(256)
void prep_bmask(const float* __restrict__ eb, const int* __restrict__ adj,
                float* __restrict__ bm) {
  int i = blockIdx.x * 256 + threadIdx.x;           // < N*N
  int n = i >> 10, m = i & (Nn - 1);
  float v = eb[i];
  if (adj[i] == 0 && n != m) v = -1e30f;            // mask non-adjacent, keep self
  bm[i] = v;
}

// ---------------------------------------------------------------- fused QKV GEMM
// C[i,j] = sum_k enh[i,k] * W[j,k] + b[j]   (A = enh tile, B = W^T tile)
__global__ __launch_bounds__(256)
void qkv_kernel(const __bf16* __restrict__ enh,
                const __bf16* __restrict__ Wqb, const __bf16* __restrict__ Wkb,
                const __bf16* __restrict__ Wvb,
                const float* __restrict__ bq, const float* __restrict__ bk,
                const float* __restrict__ bv,
                __bf16* __restrict__ Qb, __bf16* __restrict__ Kb,
                __bf16* __restrict__ Vb) {
  const int lane = threadIdx.x & 31;
  const int w    = threadIdx.x >> 5;      // wave -> column tile 0..7
  const int hl   = lane >> 4, l15 = lane & 15;
  const int row0 = blockIdx.x * 16;
  const int j    = w * 16 + l15;          // output column this lane owns
  const int rA   = row0 + l15;            // A-matrix row this lane owns

  v8f aq = {}, ak = {}, av = {};
#pragma unroll
  for (int kb = 0; kb < 4; ++kb) {
    const int k0 = kb * 32;
    // A 16x32 bf16 fragment: lane<16 holds K={0..15}, lane>=16 K={16..31}
    v8bf lo = *(const v8bf*)(enh + (size_t)rA * Dd + k0 + hl * 16);
    v8bf hi = *(const v8bf*)(enh + (size_t)rA * Dd + k0 + 8 + hl * 16);
    v16bf af;
#pragma unroll
    for (int e = 0; e < 8; ++e) { af[e] = lo[e]; af[8 + e] = hi[e]; }
    // B 32x16 fragment: lane = column j, elements = K (hl*16 + e); B[k,j] = W[j,k]
    v16bf bqf = *(const v16bf*)(Wqb + (size_t)j * Dd + k0 + hl * 16);
    v16bf bkf = *(const v16bf*)(Wkb + (size_t)j * Dd + k0 + hl * 16);
    v16bf bvf = *(const v16bf*)(Wvb + (size_t)j * Dd + k0 + hl * 16);
    aq = __builtin_amdgcn_wmma_f32_16x16x32_bf16(false, af, false, bqf, (short)0, aq, false, false);
    ak = __builtin_amdgcn_wmma_f32_16x16x32_bf16(false, af, false, bkf, (short)0, ak, false, false);
    av = __builtin_amdgcn_wmma_f32_16x16x32_bf16(false, af, false, bvf, (short)0, av, false, false);
  }
  const float cq = bq[j], ck = bk[j], cv = bv[j];
#pragma unroll
  for (int r = 0; r < 8; ++r) {
    const size_t m = (size_t)(row0 + hl * 8 + r) * Dd + j;   // D: VGPR r -> row r / 8+r
    Qb[m] = (__bf16)(aq[r] + cq);
    Kb[m] = (__bf16)(ak[r] + ck);
    Vb[m] = (__bf16)(av[r] + cv);
  }
}

// ---------------------------------------------------------------- flash attention
// S^T = K * Q^T  (queries live in lanes -> softmax is lane-local + one shfl_xor(16))
// V chunks are double-buffered in LDS via async global->LDS copies (ASYNCcnt).
__global__ __launch_bounds__(256)
void attn_kernel(const __bf16* __restrict__ Qb, const __bf16* __restrict__ Kb,
                 const __bf16* __restrict__ Vb, const float* __restrict__ bm,
                 __bf16* __restrict__ AOb) {
  __shared__ __align__(128) __bf16 ldsV[8][2][32][48];  // 96B row stride, 3072B/buffer
  const int lane = threadIdx.x & 31;
  const int w    = threadIdx.x >> 5;
  const int hl   = lane >> 4, l15 = lane & 15;
  int wid  = blockIdx.x * 8 + w;                     // 6144 waves total
  const int ntile = wid & 63;  wid >>= 6;
  const int t = wid % Tt;      wid /= Tt;
  const int h = wid % Hh;
  const int b = wid / Hh;
  const int n0 = ntile * 16;
  const float scale = 0.1767766952966369f;           // 1/sqrt(32)

  // LDS byte address of this wave's slab (LDS aperture truncates to addr[31:0])
  const unsigned ldsWave = (unsigned)(uintptr_t)&ldsV[w][0][0][0];
  constexpr unsigned BUFB = 32u * 48u * 2u;          // 3072 bytes per buffer

  // issue one 32x32 V chunk as 128 x 16B async transfers (4 per lane)
  auto issueV = [&](int m0c, int buf) {
#pragma unroll
    for (int i = 0; i < 4; ++i) {
      const int c   = i * 32 + lane;
      const int row = c >> 2, part = c & 3;
      const size_t gRow = (size_t)((b * Nn + m0c + row) * Tt + t);
      const __bf16* g = Vb + gRow * Dd + h * 32 + part * 8;
      const unsigned loff = ldsWave + (unsigned)buf * BUFB
                          + (unsigned)row * 96u + (unsigned)part * 16u;
      async_g2l_b128(loff, g);
    }
  };

  // Q^T B-fragment (32x16): lane = query col, elements = d (hl*16 + e)
  const int nq = n0 + l15;
  const size_t rowQ = (size_t)((b * Nn + nq) * Tt + t);
  const v16bf qf = *(const v16bf*)(Qb + rowQ * Dd + h * 32 + hl * 16);
  const float* bmrow = bm + (size_t)nq * Nn;

  float rmax = -1e30f, lsum = 0.f;
  v8f olo = {}, ohi = {};

  issueV(0, 0);                                      // prologue: chunk 0 in flight

  for (int ch = 0; ch < NCH; ++ch) {
    const int m0  = ch * 32;
    const int buf = ch & 1;
    const bool more = (ch + 1 < NCH);
    if (more) issueV(m0 + 32, buf ^ 1);              // overlap next V copy with compute
    if (more) {                                      // speculative prefetch of next K chunk
      const size_t pr = (size_t)((b * Nn + m0 + 32 + l15) * Tt + t);
      __builtin_prefetch(Kb + pr * Dd + h * 32, 0, 1);
    }

    // ---- K A-fragments for the two 16-key tiles (full DH=32 in one WMMA K-step)
    const size_t rk0 = (size_t)((b * Nn + m0 + l15) * Tt + t);
    const size_t rk1 = (size_t)((b * Nn + m0 + 16 + l15) * Tt + t);
    v8bf k0lo = *(const v8bf*)(Kb + rk0 * Dd + h * 32 + hl * 16);
    v8bf k0hi = *(const v8bf*)(Kb + rk0 * Dd + h * 32 + 8 + hl * 16);
    v8bf k1lo = *(const v8bf*)(Kb + rk1 * Dd + h * 32 + hl * 16);
    v8bf k1hi = *(const v8bf*)(Kb + rk1 * Dd + h * 32 + 8 + hl * 16);
    v16bf kf0, kf1;
#pragma unroll
    for (int e = 0; e < 8; ++e) {
      kf0[e] = k0lo[e]; kf0[8 + e] = k0hi[e];
      kf1[e] = k1lo[e]; kf1[8 + e] = k1hi[e];
    }
    v8f z = {};
    v8f s0 = __builtin_amdgcn_wmma_f32_16x16x32_bf16(false, kf0, false, qf, (short)0, z, false, false);
    v8f s1 = __builtin_amdgcn_wmma_f32_16x16x32_bf16(false, kf1, false, qf, (short)0, z, false, false);

    // scale + edge-bias/mask (rows m = hl*8 + r within each tile)
    v8f b0 = *(const v8f*)(bmrow + m0 + hl * 8);
    v8f b1 = *(const v8f*)(bmrow + m0 + 16 + hl * 8);
    float sc0[8], sc1[8];
    float cmax = -1e30f;
#pragma unroll
    for (int r = 0; r < 8; ++r) {
      sc0[r] = s0[r] * scale + b0[r];
      sc1[r] = s1[r] * scale + b1[r];
      cmax = fmaxf(cmax, fmaxf(sc0[r], sc1[r]));
    }
    cmax = fmaxf(cmax, __shfl_xor(cmax, 16));        // merge the two key-halves
    const float mnew = fmaxf(rmax, cmax);
    const float corr = __expf(rmax - mnew);
    rmax = mnew;

    v16bf pf;                                        // P A-fragment, layout-compatible!
    float ps = 0.f;
#pragma unroll
    for (int r = 0; r < 8; ++r) {
      float p0 = __expf(sc0[r] - mnew);
      float p1 = __expf(sc1[r] - mnew);
      ps += p0 + p1;
      pf[r] = (__bf16)p0;                            // m = {0..7 | 8..15}
      pf[8 + r] = (__bf16)p1;                        // m = {16..23 | 24..31}
    }
    ps += __shfl_xor(ps, 16);
    lsum = lsum * corr + ps;

    // rescale running O (rows = queries, broadcast per-row factor from stat lanes)
#pragma unroll
    for (int r = 0; r < 8; ++r) {
      float cr = __shfl(corr, hl ? 8 + r : r);
      olo[r] *= cr; ohi[r] *= cr;
    }

    // wait for THIS chunk's async V copy (in-order: <=4 leaves next chunk in flight)
    if (more) wait_async_le4(); else wait_async_le0();

    // V B-fragments from LDS column gather: lane = dim col, elements = keys
    v16bf vf0, vf1;
#pragma unroll
    for (int e = 0; e < 16; ++e) {
      vf0[e] = ldsV[w][buf][((e >> 3) << 4) + (hl << 3) + (e & 7)][l15];
      vf1[e] = ldsV[w][buf][((e >> 3) << 4) + (hl << 3) + (e & 7)][16 + l15];
    }
    olo = __builtin_amdgcn_wmma_f32_16x16x32_bf16(false, pf, false, vf0, (short)0, olo, false, false);
    ohi = __builtin_amdgcn_wmma_f32_16x16x32_bf16(false, pf, false, vf1, (short)0, ohi, false, false);
  }

  // final normalize + store (bf16) into pre-projection buffer
  const float lg = fmaxf(lsum, 1e-20f);
#pragma unroll
  for (int r = 0; r < 8; ++r) {
    float li = 1.f / __shfl(lg, hl ? 8 + r : r);
    const size_t rowO = (size_t)((b * Nn + n0 + hl * 8 + r) * Tt + t);
    AOb[rowO * Dd + h * 32 + l15]      = (__bf16)(olo[r] * li);
    AOb[rowO * Dd + h * 32 + 16 + l15] = (__bf16)(ohi[r] * li);
  }
}

// ---------------------------------------------------------------- Wo GEMM + residual + LayerNorm
__global__ __launch_bounds__(256)
void outln_kernel(const __bf16* __restrict__ AOb, const __bf16* __restrict__ Wob,
                  const float* __restrict__ bo, const float* __restrict__ x,
                  const float* __restrict__ g, const float* __restrict__ bb,
                  float* __restrict__ out) {
  __shared__ float ldsO[16][132];
  const int lane = threadIdx.x & 31;
  const int w    = threadIdx.x >> 5;
  const int hl   = lane >> 4, l15 = lane & 15;
  const int row0 = blockIdx.x * 16;
  const int j    = w * 16 + l15;
  const int rA   = row0 + l15;

  v8f acc = {};
#pragma unroll
  for (int kb = 0; kb < 4; ++kb) {
    const int k0 = kb * 32;
    v8bf lo = *(const v8bf*)(AOb + (size_t)rA * Dd + k0 + hl * 16);
    v8bf hi = *(const v8bf*)(AOb + (size_t)rA * Dd + k0 + 8 + hl * 16);
    v16bf af;
#pragma unroll
    for (int e = 0; e < 8; ++e) { af[e] = lo[e]; af[8 + e] = hi[e]; }
    v16bf bf = *(const v16bf*)(Wob + (size_t)j * Dd + k0 + hl * 16);
    acc = __builtin_amdgcn_wmma_f32_16x16x32_bf16(false, af, false, bf, (short)0, acc, false, false);
  }
  const float bj = bo[j];
#pragma unroll
  for (int r = 0; r < 8; ++r) ldsO[hl * 8 + r][j] = acc[r] + bj;
  __syncthreads();

  // LayerNorm: 16 lanes per row, 8 elems each, shfl-reduce within half-wave
  const int rr = threadIdx.x >> 4, part = threadIdx.x & 15;
  const size_t grow = (size_t)(row0 + rr);
  float v[8], s = 0.f, ss = 0.f;
#pragma unroll
  for (int i = 0; i < 8; ++i) {
    const int c = part * 8 + i;
    float val = ldsO[rr][c] + x[grow * Dd + c];      // + residual
    v[i] = val; s += val; ss += val * val;
  }
#pragma unroll
  for (int m = 1; m <= 8; m <<= 1) { s += __shfl_xor(s, m); ss += __shfl_xor(ss, m); }
  const float mean = s * (1.f / Dd);
  const float var  = ss * (1.f / Dd) - mean * mean;
  const float rs   = rsqrtf(var + 1e-5f);
#pragma unroll
  for (int i = 0; i < 8; ++i) {
    const int c = part * 8 + i;
    out[grow * Dd + c] = (v[i] - mean) * rs * g[c] + bb[c];
  }
}

// ---------------------------------------------------------------- launch
extern "C" void kernel_launch(void* const* d_in, const int* in_sizes, int n_in,
                              void* d_out, int out_size, void* d_ws, size_t ws_size,
                              hipStream_t stream) {
  (void)in_sizes; (void)n_in; (void)out_size; (void)ws_size;
  const float* x   = (const float*)d_in[0];
  const int*   adj = (const int*)  d_in[1];
  const float* emb = (const float*)d_in[2];
  const float* Wq  = (const float*)d_in[3];
  const float* bq  = (const float*)d_in[4];
  const float* Wk  = (const float*)d_in[5];
  const float* bk  = (const float*)d_in[6];
  const float* Wv  = (const float*)d_in[7];
  const float* bv  = (const float*)d_in[8];
  const float* eb  = (const float*)d_in[9];
  const float* Wo  = (const float*)d_in[10];
  const float* bo  = (const float*)d_in[11];
  const float* lg  = (const float*)d_in[12];
  const float* lb  = (const float*)d_in[13];
  float* out = (float*)d_out;

  char* p = (char*)d_ws;
  __bf16* enh = (__bf16*)p; p += (size_t)ROWS * Dd * 2;
  __bf16* Qb  = (__bf16*)p; p += (size_t)ROWS * Dd * 2;
  __bf16* Kb  = (__bf16*)p; p += (size_t)ROWS * Dd * 2;
  __bf16* Vb  = (__bf16*)p; p += (size_t)ROWS * Dd * 2;
  __bf16* AOb = (__bf16*)p; p += (size_t)ROWS * Dd * 2;
  __bf16* Wqb = (__bf16*)p; p += (size_t)WMAT * 2;
  __bf16* Wkb = (__bf16*)p; p += (size_t)WMAT * 2;
  __bf16* Wvb = (__bf16*)p; p += (size_t)WMAT * 2;
  __bf16* Wob = (__bf16*)p; p += (size_t)WMAT * 2;
  float*  bms = (float*)p;  p += (size_t)Nn * Nn * 4;

  prep_weights<<<WMAT / 256, 256, 0, stream>>>(Wq, Wk, Wv, Wo, Wqb, Wkb, Wvb, Wob);
  prep_enh<<<(ROWS * Dd) / 256, 256, 0, stream>>>(x, emb, enh);
  prep_bmask<<<(Nn * Nn) / 256, 256, 0, stream>>>(eb, adj, bms);
  qkv_kernel<<<ROWS / 16, 256, 0, stream>>>(enh, Wqb, Wkb, Wvb, bq, bk, bv, Qb, Kb, Vb);
  attn_kernel<<<(Bb * Hh * Tt * (Nn / 16)) / 8, 256, 0, stream>>>(Qb, Kb, Vb, bms, AOb);
  outln_kernel<<<ROWS / 16, 256, 0, stream>>>(AOb, Wob, bo, x, lg, lb, out);
}
